// GATv2GCN22_38560216384100
// MI455X (gfx1250) — matrix operand order repaired and checked by reading, not compiled, not run on hardware
//
#include <hip/hip_runtime.h>
#include <cstdint>
#include <cstddef>

// Problem constants (match reference)
#define NN   20000
#define EE   320000
#define RR   4
#define HH   4
#define HIDD 64
#define DINN 256
#define OUTD 64
#define HDIM 256   // H*HID
#define NEG_SLOPE 0.2f

typedef __attribute__((ext_vector_type(16))) __bf16 v16bf;
typedef __attribute__((ext_vector_type(8)))  float v8f;

union FragAB  { unsigned u[8]; v16bf v; };
union FragAcc { float   f[8]; v8f  v; };

#define GF_ACCUM 1
#define GF_RELU  2
#define GF_BIAS  4

// ---------- small helpers ----------
__device__ __forceinline__ unsigned short f2bf(float f) {
    unsigned u = __float_as_uint(f);
    unsigned r = u + 0x7FFFu + ((u >> 16) & 1u);   // RNE
    return (unsigned short)(r >> 16);
}
__device__ __forceinline__ unsigned encF(float x) {  // monotone f32->u32
    unsigned u = __float_as_uint(x);
    return (u & 0x80000000u) ? ~u : (u | 0x80000000u);
}
__device__ __forceinline__ float decF(unsigned s) {
    return __uint_as_float((s & 0x80000000u) ? (s ^ 0x80000000u) : ~s);
}
__device__ __forceinline__ float lrdot(float l, float r, float a) {
    float z = l + r;
    z = z > 0.f ? z : NEG_SLOPE * z;
    return z * a;
}

// ---------- utility kernels ----------
__global__ void fill_u32_kernel(unsigned* __restrict__ p, unsigned v, int n) {
    int i = blockIdx.x * blockDim.x + threadIdx.x;
    if (i < n) p[i] = v;
}

__global__ void tobf16_kernel(const float* __restrict__ in,
                              unsigned short* __restrict__ out, int n) {
    int i = blockIdx.x * blockDim.x + threadIdx.x;
    if (i < n) out[i] = f2bf(in[i]);
}

__global__ void bias_relu_bf16_kernel(const float* __restrict__ acc,
                                      const float* __restrict__ bias,
                                      float* __restrict__ hF,
                                      unsigned short* __restrict__ hB,
                                      int n, int widthMask) {
    int i = blockIdx.x * blockDim.x + threadIdx.x;
    if (i >= n) return;
    float v = acc[i] + bias[i & widthMask];
    v = v > 0.f ? v : 0.f;
    hF[i] = v;
    hB[i] = f2bf(v);
}

// Pack f32 weight [K][Ncol] into bf16 B-fragment order:
// out[ ((kt*NT + nt)*32 + lane)*8 + v ] = {K=kt*32+(lane&16)+2v, col=nt*16+(lane&15)} pair
__global__ void pack_w_kernel(const float* __restrict__ W,
                              unsigned* __restrict__ out, int K, int Ncol) {
    int i = blockIdx.x * blockDim.x + threadIdx.x;
    int total = (K >> 5) * (Ncol >> 4) * 256;
    if (i >= total) return;
    int v    = i & 7;
    int lane = (i >> 3) & 31;
    int tile = i >> 8;
    int NT   = Ncol >> 4;
    int nt   = tile % NT;
    int kt   = tile / NT;
    int k0   = kt * 32 + (lane & 16) + (v << 1);
    int col  = nt * 16 + (lane & 15);
    unsigned lo = f2bf(W[(size_t)k0 * Ncol + col]);
    unsigned hi = f2bf(W[(size_t)(k0 + 1) * Ncol + col]);
    out[i] = lo | (hi << 16);
}

// ---------- WMMA GEMM: C[M,Ncol](ldc) = A[M,K](bf16) @ Wp + bias ----------
__global__ __launch_bounds__(128) void gemm_bf16_wmma(
    const unsigned short* __restrict__ A, const unsigned* __restrict__ Wp,
    const float* __restrict__ bias, float* __restrict__ C,
    int M, int K, int Ncol, int ldc, int flags)
{
    __shared__ unsigned short tileA[64][34];   // padded: 17 dwords/row, conflict-free
    const int lane  = threadIdx.x & 31;
    const int wave  = threadIdx.x >> 5;
    const int rBase = blockIdx.x * 64 + wave * 16;
    const int cBase = blockIdx.y * 64;
    const int NT    = Ncol >> 4;

    FragAcc acc[4];
    for (int t = 0; t < 4; ++t)
        for (int j = 0; j < 8; ++j) acc[t].f[j] = 0.0f;

    const int lrow  = threadIdx.x >> 1;
    const int lhalf = threadIdx.x & 1;
    int rowg = blockIdx.x * 64 + lrow;
    if (rowg >= M) rowg = M - 1;               // clamp (stores are guarded)
    const int arow = wave * 16 + (lane & 15);
    const int kHiA = (lane >> 4) << 3;

    const int nkt = K >> 5;
    for (int kt = 0; kt < nkt; ++kt) {
        // stage 64x32 bf16 A-tile into LDS (coalesced 2x16B per thread)
        const unsigned short* ap = A + (size_t)rowg * K + kt * 32 + lhalf * 16;
        uint4 d0 = *reinterpret_cast<const uint4*>(ap);
        uint4 d1 = *reinterpret_cast<const uint4*>(ap + 8);
        unsigned* lr = reinterpret_cast<unsigned*>(&tileA[lrow][lhalf * 16]);
        lr[0] = d0.x; lr[1] = d0.y; lr[2] = d0.z; lr[3] = d0.w;
        lr[4] = d1.x; lr[5] = d1.y; lr[6] = d1.z; lr[7] = d1.w;
        __syncthreads();

        // A fragment (ISA 7.12.2: 16-bit A 16x32, K-pairs across 8 VGPRs)
        FragAB af;
        #pragma unroll
        for (int v = 0; v < 8; ++v) {
            int k0 = (kHiA << 1) + (v << 1);
            af.u[v] = *reinterpret_cast<const unsigned*>(&tileA[arow][k0]);
        }
        // 4 B fragments (pre-packed, contiguous 32B/lane) + WMMA
        #pragma unroll
        for (int t = 0; t < 4; ++t) {
            const unsigned* bp =
                Wp + (((size_t)(kt * NT + (cBase >> 4) + t)) * 32 + lane) * 8;
            uint4 b0 = *reinterpret_cast<const uint4*>(bp);
            uint4 b1 = *reinterpret_cast<const uint4*>(bp + 4);
            FragAB bfr;
            bfr.u[0] = b0.x; bfr.u[1] = b0.y; bfr.u[2] = b0.z; bfr.u[3] = b0.w;
            bfr.u[4] = b1.x; bfr.u[5] = b1.y; bfr.u[6] = b1.z; bfr.u[7] = b1.w;
            acc[t].v = __builtin_amdgcn_wmma_f32_16x16x32_bf16(
                false, af.v, false, bfr.v, (short)0, acc[t].v, false, false);
        }
        __syncthreads();
    }

    // epilogue: D layout (VGPR r -> M=r / r+8 by lane half; N = lane&15)
    const int mrow0 = rBase + ((lane >> 4) << 3);
    const int ncol0 = cBase + (lane & 15);
    for (int t = 0; t < 4; ++t) {
        int col = ncol0 + t * 16;
        float b = (flags & GF_BIAS) ? bias[col] : 0.0f;
        #pragma unroll
        for (int r = 0; r < 8; ++r) {
            int m = mrow0 + r;
            if (m < M) {
                float v = acc[t].f[r] + b;
                float* cp = C + (size_t)m * ldc + col;
                if (flags & GF_ACCUM) v += *cp;
                if (flags & GF_RELU)  v = v > 0.f ? v : 0.f;
                *cp = v;
            }
        }
    }
}

// ---------- GATv2 edge kernels (wave-per-edge, self loops implicit for eid>=E) ----------
__global__ __launch_bounds__(256) void gat_edge_logits(
    const float* __restrict__ xl, const float* __restrict__ xr,
    const int* __restrict__ src, const int* __restrict__ dst,
    const float* __restrict__ att, float* __restrict__ eBuf,
    unsigned* __restrict__ mU, int E, int n)
{
    int gid  = blockIdx.x * blockDim.x + threadIdx.x;
    int eid  = gid >> 5;
    int lane = gid & 31;
    if (eid >= E + n) return;
    int s = (eid < E) ? src[eid] : (eid - E);
    int d = (eid < E) ? dst[eid] : (eid - E);

    const float4* pl = reinterpret_cast<const float4*>(xl + (size_t)s * HDIM + lane * 8);
    const float4* pr = reinterpret_cast<const float4*>(xr + (size_t)d * HDIM + lane * 8);
    const float4* pa = reinterpret_cast<const float4*>(att + lane * 8);
    float4 l0 = pl[0], l1 = pl[1];
    float4 r0 = pr[0], r1 = pr[1];
    float4 a0 = pa[0], a1 = pa[1];

    float p = 0.f;
    p += lrdot(l0.x, r0.x, a0.x); p += lrdot(l0.y, r0.y, a0.y);
    p += lrdot(l0.z, r0.z, a0.z); p += lrdot(l0.w, r0.w, a0.w);
    p += lrdot(l1.x, r1.x, a1.x); p += lrdot(l1.y, r1.y, a1.y);
    p += lrdot(l1.z, r1.z, a1.z); p += lrdot(l1.w, r1.w, a1.w);

    // reduce within 8-lane head groups
    p += __shfl_xor(p, 1);
    p += __shfl_xor(p, 2);
    p += __shfl_xor(p, 4);
    if ((lane & 7) == 0) {
        int h = lane >> 3;
        eBuf[(size_t)eid * HH + h] = p;
        atomicMax(&mU[(size_t)d * HH + h], encF(p));
    }
}

__global__ void gat_edge_softmax_num(
    float* __restrict__ eBuf, const unsigned* __restrict__ mU,
    float* __restrict__ sBuf, const int* __restrict__ dst, int E, int n)
{
    int t = blockIdx.x * blockDim.x + threadIdx.x;
    if (t >= (E + n) * HH) return;
    int eid = t >> 2, h = t & 3;
    int d = (eid < E) ? dst[eid] : (eid - E);
    float m = decF(mU[(size_t)d * HH + h]);
    float p = __expf(eBuf[t] - m);
    eBuf[t] = p;
    atomicAdd(&sBuf[(size_t)d * HH + h], p);
}

__global__ __launch_bounds__(256) void gat_edge_aggregate(
    const float* __restrict__ eBuf, const float* __restrict__ sBuf,
    const float* __restrict__ xl, const int* __restrict__ src,
    const int* __restrict__ dst, float* __restrict__ outAcc, int E, int n)
{
    int gid  = blockIdx.x * blockDim.x + threadIdx.x;
    int eid  = gid >> 5;
    int lane = gid & 31;
    if (eid >= E + n) return;
    int s = (eid < E) ? src[eid] : (eid - E);
    int d = (eid < E) ? dst[eid] : (eid - E);
    int h = lane >> 3;
    float alpha = eBuf[(size_t)eid * HH + h] / sBuf[(size_t)d * HH + h];

    const float4* ps = reinterpret_cast<const float4*>(xl + (size_t)s * HDIM + lane * 8);
    float4 v0 = ps[0], v1 = ps[1];
    float* po = outAcc + (size_t)d * HDIM + lane * 8;
    atomicAdd(po + 0, alpha * v0.x); atomicAdd(po + 1, alpha * v0.y);
    atomicAdd(po + 2, alpha * v0.z); atomicAdd(po + 3, alpha * v0.w);
    atomicAdd(po + 4, alpha * v1.x); atomicAdd(po + 5, alpha * v1.y);
    atomicAdd(po + 6, alpha * v1.z); atomicAdd(po + 7, alpha * v1.w);
}

// ---------- GraphConv aggregation (no self loops) ----------
__global__ __launch_bounds__(256) void graph_agg_kernel(
    const float* __restrict__ x, const int* __restrict__ src,
    const int* __restrict__ dst, float* __restrict__ agg, int E, int width)
{
    int gid  = blockIdx.x * blockDim.x + threadIdx.x;
    int eid  = gid >> 5;
    int lane = gid & 31;
    if (eid >= E) return;
    int s = src[eid], d = dst[eid];
    for (int c = lane * 4; c < width; c += 128) {
        float4 v = *reinterpret_cast<const float4*>(x + (size_t)s * width + c);
        float* p = agg + (size_t)d * width + c;
        atomicAdd(p + 0, v.x); atomicAdd(p + 1, v.y);
        atomicAdd(p + 2, v.z); atomicAdd(p + 3, v.w);
    }
}

// ---------- host side ----------
static inline size_t alignUp(size_t x, size_t a) { return (x + a - 1) & ~(a - 1); }

extern "C" void kernel_launch(void* const* d_in, const int* in_sizes, int n_in,
                              void* d_out, int out_size, void* d_ws, size_t ws_size,
                              hipStream_t stream) {
    const float* x     = (const float*)d_in[0];
    const int*   ei    = (const int*)  d_in[1];
    const float* Wl1   = (const float*)d_in[2];
    const float* Wr1   = (const float*)d_in[3];
    const float* att1  = (const float*)d_in[4];
    const float* b1    = (const float*)d_in[5];
    const float* Wl2   = (const float*)d_in[6];
    const float* Wr2   = (const float*)d_in[7];
    const float* att2  = (const float*)d_in[8];
    const float* b2    = (const float*)d_in[9];
    const float* Wrel3 = (const float*)d_in[10];
    const float* Wroot3= (const float*)d_in[11];
    const float* b3    = (const float*)d_in[12];
    const float* Wrel4 = (const float*)d_in[13];
    const float* Wroot4= (const float*)d_in[14];
    const float* b4    = (const float*)d_in[15];
    float* outp = (float*)d_out;

    char* ws = (char*)d_ws;
    size_t off = 0;
    auto take = [&](size_t bytes) -> void* {
        void* p = ws + off;
        off = alignUp(off + bytes, 256);
        return p;
    };

    unsigned short* xbf   = (unsigned short*)take((size_t)NN * HDIM * 2);
    unsigned short* hB    = (unsigned short*)take((size_t)NN * HDIM * 2);
    unsigned short* aggB  = (unsigned short*)take((size_t)NN * HDIM * 2);
    float* xl   = (float*)take((size_t)NN * HDIM * 4);
    float* xr   = (float*)take((size_t)NN * HDIM * 4);
    float* accO = (float*)take((size_t)NN * HDIM * 4);
    float* hF   = (float*)take((size_t)NN * HDIM * 4);
    float* eBuf = (float*)take((size_t)(EE + NN) * HH * 4);
    float* sBuf = (float*)take((size_t)NN * HH * 4);
    unsigned* mU = (unsigned*)take((size_t)NN * HH * 4);
    float* h3F  = (float*)take((size_t)NN * OUTD * 4);
    unsigned short* h3B   = (unsigned short*)take((size_t)NN * OUTD * 2);
    float* agg4F = (float*)take((size_t)NN * OUTD * 4);
    unsigned short* agg4B = (unsigned short*)take((size_t)NN * OUTD * 2);
    unsigned* wp0 = (unsigned*)take((size_t)DINN * HDIM / 2 * 4);
    unsigned* wp1 = (unsigned*)take((size_t)DINN * HDIM / 2 * 4);

    const int EA = EE + NN;
    const int edgeBlocks = (EA * 32 + 255) / 256;   // wave per edge incl self loops
    const int eBlocks    = (EE * 32 + 255) / 256;   // wave per raw edge
    const dim3 gemmBig((NN + 63) / 64, HDIM / 64);  // Ncol=256
    const dim3 gemmSm ((NN + 63) / 64, OUTD / 64);  // Ncol=64
    const unsigned NEG_INF_ENC = 0x007FFFFFu;       // encF(-inf)

    auto ceil256 = [](int n) { return (n + 255) / 256; };

    // x -> bf16 once
    tobf16_kernel<<<ceil256(NN * HDIM), 256, 0, stream>>>(x, xbf, NN * HDIM);

    for (int r = 0; r < RR; ++r) {
        const int* srcP = ei + (size_t)(2 * r    ) * EE;
        const int* dstP = ei + (size_t)(2 * r + 1) * EE;

        // ---------- Layer 1: GATv2 on x ----------
        pack_w_kernel<<<ceil256(DINN * HDIM / 2), 256, 0, stream>>>(Wl1 + (size_t)r * DINN * HDIM, wp0, DINN, HDIM);
        pack_w_kernel<<<ceil256(DINN * HDIM / 2), 256, 0, stream>>>(Wr1 + (size_t)r * DINN * HDIM, wp1, DINN, HDIM);
        gemm_bf16_wmma<<<gemmBig, 128, 0, stream>>>(xbf, wp0, nullptr, xl, NN, DINN, HDIM, HDIM, 0);
        gemm_bf16_wmma<<<gemmBig, 128, 0, stream>>>(xbf, wp1, nullptr, xr, NN, DINN, HDIM, HDIM, 0);
        fill_u32_kernel<<<ceil256(NN * HH), 256, 0, stream>>>(mU, NEG_INF_ENC, NN * HH);
        fill_u32_kernel<<<ceil256(NN * HH), 256, 0, stream>>>((unsigned*)sBuf, 0u, NN * HH);
        fill_u32_kernel<<<ceil256(NN * HDIM), 256, 0, stream>>>((unsigned*)accO, 0u, NN * HDIM);
        gat_edge_logits<<<edgeBlocks, 256, 0, stream>>>(xl, xr, srcP, dstP, att1 + (size_t)r * HDIM, eBuf, mU, EE, NN);
        gat_edge_softmax_num<<<ceil256(EA * HH), 256, 0, stream>>>(eBuf, mU, sBuf, dstP, EE, NN);
        gat_edge_aggregate<<<edgeBlocks, 256, 0, stream>>>(eBuf, sBuf, xl, srcP, dstP, accO, EE, NN);
        bias_relu_bf16_kernel<<<ceil256(NN * HDIM), 256, 0, stream>>>(accO, b1 + (size_t)r * HDIM, hF, hB, NN * HDIM, HDIM - 1);

        // ---------- Layer 2: GATv2 on h1 ----------
        pack_w_kernel<<<ceil256(HDIM * HDIM / 2), 256, 0, stream>>>(Wl2 + (size_t)r * HDIM * HDIM, wp0, HDIM, HDIM);
        pack_w_kernel<<<ceil256(HDIM * HDIM / 2), 256, 0, stream>>>(Wr2 + (size_t)r * HDIM * HDIM, wp1, HDIM, HDIM);
        gemm_bf16_wmma<<<gemmBig, 128, 0, stream>>>(hB, wp0, nullptr, xl, NN, HDIM, HDIM, HDIM, 0);
        gemm_bf16_wmma<<<gemmBig, 128, 0, stream>>>(hB, wp1, nullptr, xr, NN, HDIM, HDIM, HDIM, 0);
        fill_u32_kernel<<<ceil256(NN * HH), 256, 0, stream>>>(mU, NEG_INF_ENC, NN * HH);
        fill_u32_kernel<<<ceil256(NN * HH), 256, 0, stream>>>((unsigned*)sBuf, 0u, NN * HH);
        fill_u32_kernel<<<ceil256(NN * HDIM), 256, 0, stream>>>((unsigned*)accO, 0u, NN * HDIM);
        gat_edge_logits<<<edgeBlocks, 256, 0, stream>>>(xl, xr, srcP, dstP, att2 + (size_t)r * HDIM, eBuf, mU, EE, NN);
        gat_edge_softmax_num<<<ceil256(EA * HH), 256, 0, stream>>>(eBuf, mU, sBuf, dstP, EE, NN);
        gat_edge_aggregate<<<edgeBlocks, 256, 0, stream>>>(eBuf, sBuf, xl, srcP, dstP, accO, EE, NN);
        bias_relu_bf16_kernel<<<ceil256(NN * HDIM), 256, 0, stream>>>(accO, b2 + (size_t)r * HDIM, hF, hB, NN * HDIM, HDIM - 1);

        // ---------- Layer 3: GraphConv 256 -> 64, ReLU ----------
        fill_u32_kernel<<<ceil256(NN * HDIM), 256, 0, stream>>>((unsigned*)accO, 0u, NN * HDIM);
        graph_agg_kernel<<<eBlocks, 256, 0, stream>>>(hF, srcP, dstP, accO, EE, HDIM);
        tobf16_kernel<<<ceil256(NN * HDIM), 256, 0, stream>>>(accO, aggB, NN * HDIM);
        pack_w_kernel<<<ceil256(HDIM * OUTD / 2), 256, 0, stream>>>(Wrel3  + (size_t)r * HDIM * OUTD, wp0, HDIM, OUTD);
        pack_w_kernel<<<ceil256(HDIM * OUTD / 2), 256, 0, stream>>>(Wroot3 + (size_t)r * HDIM * OUTD, wp1, HDIM, OUTD);
        gemm_bf16_wmma<<<gemmSm, 128, 0, stream>>>(aggB, wp0, b3 + (size_t)r * OUTD, h3F, NN, HDIM, OUTD, OUTD, GF_BIAS);
        gemm_bf16_wmma<<<gemmSm, 128, 0, stream>>>(hB,   wp1, nullptr,               h3F, NN, HDIM, OUTD, OUTD, GF_ACCUM | GF_RELU);
        tobf16_kernel<<<ceil256(NN * OUTD), 256, 0, stream>>>(h3F, h3B, NN * OUTD);

        // ---------- Layer 4: GraphConv 64 -> 64, strided into d_out[N,R,OUT] ----------
        fill_u32_kernel<<<ceil256(NN * OUTD), 256, 0, stream>>>((unsigned*)agg4F, 0u, NN * OUTD);
        graph_agg_kernel<<<eBlocks, 256, 0, stream>>>(h3F, srcP, dstP, agg4F, EE, OUTD);
        tobf16_kernel<<<ceil256(NN * OUTD), 256, 0, stream>>>(agg4F, agg4B, NN * OUTD);
        pack_w_kernel<<<ceil256(OUTD * OUTD / 2), 256, 0, stream>>>(Wrel4  + (size_t)r * OUTD * OUTD, wp0, OUTD, OUTD);
        pack_w_kernel<<<ceil256(OUTD * OUTD / 2), 256, 0, stream>>>(Wroot4 + (size_t)r * OUTD * OUTD, wp1, OUTD, OUTD);
        gemm_bf16_wmma<<<gemmSm, 128, 0, stream>>>(agg4B, wp0, b4 + (size_t)r * OUTD, outp + (size_t)r * OUTD, NN, OUTD, OUTD, RR * OUTD, GF_BIAS);
        gemm_bf16_wmma<<<gemmSm, 128, 0, stream>>>(h3B,   wp1, nullptr,               outp + (size_t)r * OUTD, NN, OUTD, OUTD, RR * OUTD, GF_ACCUM);
    }
}
